// SelfAttention_807453852004
// MI455X (gfx1250) — matrix operand order repaired and checked by reading, not compiled, not run on hardware
//
#include <hip/hip_runtime.h>
#include <hip/hip_bf16.h>

// ---------------------------------------------------------------------------
// Problem constants (B, C, H, W) = (8, 512, 32, 32); N = 1024; 8 heads, d=64.
// ---------------------------------------------------------------------------
#define BB   8
#define CC   512
#define NN   1024
#define NH   8
#define DD   64
#define C3   1536   // 3*C

typedef __attribute__((ext_vector_type(16))) _Float16      v16h;
typedef __attribute__((ext_vector_type(8)))  float         v8f;
typedef __attribute__((ext_vector_type(4)))  unsigned int  u32x4;
typedef __attribute__((ext_vector_type(8)))  int           i32x8;
typedef __attribute__((ext_vector_type(4)))  int           i32x4;

#if defined(__gfx1250__) && __has_builtin(__builtin_amdgcn_tensor_load_to_lds)
#define USE_TDM 1
#else
#define USE_TDM 0
#endif

__device__ __forceinline__ v8f wmma16x16x32(v16h a, v16h b, v8f c) {
  // D = A(16x32 f16) x B(32x16 f16) + C(16x16 f32)
  return __builtin_amdgcn_wmma_f32_16x16x32_f16(
      /*neg_a=*/false, a, /*neg_b=*/false, b,
      /*c_mod=*/(short)0, c, /*reuse_a=*/false, /*reuse_b=*/false);
}

// A-fragment: 16x32 (MxK) f16, row-major source, row stride `ld` elements.
// ISA 7.12.2: lane -> row lane%16; per lane two contiguous 16B chunks at
// k-offsets 8*half and 16+8*half.
__device__ __forceinline__ v16h load_frag_a(const _Float16* src, int ld) {
  int l = threadIdx.x & 31;
  int row = l & 15, half = l >> 4;
  const _Float16* p = src + (size_t)row * ld + 8 * half;
  union { v16h v; uint4 q[2]; } u;
  u.q[0] = *(const uint4*)(p);
  u.q[1] = *(const uint4*)(p + 16);
  return u.v;
}

// B-fragment: 32x16 (KxN) f16 where element (k, col) = base[col*ldc + k].
// lane -> col lane%16; two 16B chunks at k-offsets 8*half and 16+8*half.
__device__ __forceinline__ v16h load_frag_b_kcontig(const _Float16* base, int ldc) {
  int l = threadIdx.x & 31;
  int col = l & 15, half = l >> 4;
  const _Float16* p = base + (size_t)col * ldc + 8 * half;
  union { v16h v; uint4 q[2]; } u;
  u.q[0] = *(const uint4*)(p);
  u.q[1] = *(const uint4*)(p + 16);
  return u.v;
}

#if USE_TDM
// Issue a 2-D TDM tile load: tile_d1 rows of tile_d0 f16 elements,
// row stride stride0 elements, into LDS at lds_addr. ISA ch.8 D# bitfields.
__device__ __forceinline__ void tdm_load_2d_f16(unsigned lds_addr, const void* gaddr,
                                                unsigned tensor_d0, unsigned tensor_d1,
                                                unsigned tile_d0, unsigned tile_d1,
                                                unsigned stride0) {
  unsigned long long ga = (unsigned long long)(size_t)gaddr;
  u32x4 g0;
  g0[0] = 1u;                                               // count=1, user D#
  g0[1] = lds_addr;                                         // LDS byte address
  g0[2] = (unsigned)(ga & 0xffffffffu);                     // global_addr[31:0]
  g0[3] = (unsigned)((ga >> 32) & 0x01ffffffu) | (2u << 30);// addr[56:32] | type=2
  i32x8 g1;
  g1[0] = (int)(1u << 16);                                  // data_size=1 (2B)
  g1[1] = (int)((tensor_d0 & 0xffffu) << 16);               // tensor_dim0[15:0]
  g1[2] = (int)((tensor_d0 >> 16) | ((tensor_d1 & 0xffffu) << 16));
  g1[3] = (int)((tensor_d1 >> 16) | (tile_d0 << 16));       // tile_dim0
  g1[4] = (int)(tile_d1 & 0xffffu);                         // tile_dim1, tile_dim2=0
  g1[5] = (int)stride0;                                     // dim0_stride[31:0]
  g1[6] = 0;
  g1[7] = 0;
  i32x4 z4 = {0, 0, 0, 0};
#if __clang_major__ >= 23
  i32x8 z8 = {0, 0, 0, 0, 0, 0, 0, 0};
  __builtin_amdgcn_tensor_load_to_lds(g0, g1, z4, z4, z8, 0);
#else
  __builtin_amdgcn_tensor_load_to_lds(g0, g1, z4, z4, 0);
#endif
}
#endif

// ---------------------------------------------------------------------------
// K0: f32 -> f16 weight conversion
// ---------------------------------------------------------------------------
__global__ __launch_bounds__(256) void cvt_f16_kernel(const float* __restrict__ src,
                                                      _Float16* __restrict__ dst, int n) {
  int i = blockIdx.x * 256 + threadIdx.x;
  if (i < n) dst[i] = (_Float16)src[i];
}

// ---------------------------------------------------------------------------
// K1: transpose [B,C,N]->[B,N,C], LayerNorm over C, cast to f16.
// ---------------------------------------------------------------------------
__global__ __launch_bounds__(256) void ln1_kernel(const float* __restrict__ x,
                                                  const float* __restrict__ w,
                                                  const float* __restrict__ bias,
                                                  _Float16* __restrict__ xn) {
  int bn = blockIdx.x;
  int b = bn >> 10, n = bn & (NN - 1);
  int tid = threadIdx.x;
  const float* xp = x + (size_t)b * CC * NN + n;
  float v0 = xp[(size_t)tid * NN];
  float v1 = xp[(size_t)(tid + 256) * NN];

  __shared__ float s1[256], s2[256];
  s1[tid] = v0 + v1;
  s2[tid] = v0 * v0 + v1 * v1;
  __syncthreads();
#pragma unroll
  for (int off = 128; off > 0; off >>= 1) {
    if (tid < off) { s1[tid] += s1[tid + off]; s2[tid] += s2[tid + off]; }
    __syncthreads();
  }
  float mean = s1[0] * (1.0f / CC);
  float var  = s2[0] * (1.0f / CC) - mean * mean;
  float rstd = rsqrtf(var + 1e-5f);

  _Float16* op = xn + ((size_t)b * NN + n) * CC;
  op[tid]       = (_Float16)((v0 - mean) * rstd * w[tid]       + bias[tid]);
  op[tid + 256] = (_Float16)((v1 - mean) * rstd * w[tid + 256] + bias[tid + 256]);
}

// ---------------------------------------------------------------------------
// Shared GEMM body: 8 waves share one 16-row A tile staged once in LDS
// (k-loop invariant => single barrier), each wave owns a 16x16 output tile.
// ---------------------------------------------------------------------------
template <bool STORE_QKV>
__device__ __forceinline__ void gemm_core(const _Float16* __restrict__ Aglob,
                                          const _Float16* __restrict__ w,
                                          const float* __restrict__ bias,
                                          _Float16* __restrict__ qkv,
                                          float* __restrict__ proj,
                                          int b, int n0, int m0) {
  __shared__ __align__(16) _Float16 Ash[16 * CC];   // 16KB

  int tid = threadIdx.x;
#pragma unroll
  for (int i = 0; i < 4; ++i) {
    int idx = (tid + i * 256) * 8;                  // 8 f16 per uint4
    *(uint4*)&Ash[idx] = *(const uint4*)(Aglob + idx);
  }
  __syncthreads();

  int l = tid & 31;
  int col = l & 15, half = l >> 4;
  const _Float16* W = w + (size_t)m0 * CC;

  v8f acc;
  float bv = bias[m0 + col];
#pragma unroll
  for (int j = 0; j < 8; ++j) acc[j] = bv;

#pragma unroll 4
  for (int ks = 0; ks < CC / 32; ++ks) {
    __builtin_prefetch(W + 32 * ks + 256, 0, 1);    // global_prefetch_b8
    v16h a = load_frag_a(Ash + 32 * ks, CC);        // ds_load_b128 x2
    v16h bb = load_frag_b_kcontig(W + 32 * ks, CC); // global_load_b128 x2
    acc = wmma16x16x32(a, bb, acc);
  }

  if (STORE_QKV) {
    // scatter m -> (sel, head, dd); 16-col tiles never straddle a 64 boundary
    int sel = m0 >> 9;
    int cm = m0 & (CC - 1);
    int hh = cm >> 6, dd0 = cm & (DD - 1);
    _Float16* out = qkv + (((size_t)sel * BB + b) * NH + hh) * (NN * DD) + dd0 + col;
#pragma unroll
    for (int j = 0; j < 8; ++j) out[(size_t)(n0 + j + 8 * half) * DD] = (_Float16)acc[j];
  } else {
    float* out = proj + ((size_t)b * NN + n0) * CC + m0 + col;
#pragma unroll
    for (int j = 0; j < 8; ++j) out[(size_t)(j + 8 * half) * CC] = acc[j];
  }
}

__global__ __launch_bounds__(256) void qkv_gemm_kernel(const _Float16* __restrict__ xn,
                                                       const _Float16* __restrict__ w,
                                                       const float* __restrict__ bias,
                                                       _Float16* __restrict__ qkv) {
  int wave = threadIdx.x >> 5;
  int b = blockIdx.z, n0 = blockIdx.y * 16;
  int m0 = (blockIdx.x * 8 + wave) * 16;
  gemm_core<true>(xn + ((size_t)b * NN + n0) * CC, w, bias, qkv, nullptr, b, n0, m0);
}

__global__ __launch_bounds__(256) void out_gemm_kernel(const _Float16* __restrict__ attn,
                                                       const _Float16* __restrict__ w,
                                                       const float* __restrict__ bias,
                                                       float* __restrict__ proj) {
  int wave = threadIdx.x >> 5;
  int b = blockIdx.z, n0 = blockIdx.y * 16;
  int m0 = (blockIdx.x * 8 + wave) * 16;
  gemm_core<false>(attn + ((size_t)b * NN + n0) * CC, w, bias, nullptr, proj, b, n0, m0);
}

// ---------------------------------------------------------------------------
// K3: flash attention. All 8 waves of a block share one (b,head): the 32-key
// K tile is DMA'd into LDS by the Tensor Data Mover (wave 0 issues the D#,
// waits s_wait_tensorcnt, barrier publishes); the V tile is staged TRANSPOSED
// ([d][key]) so its B-fragment read is two contiguous ds_load_b128.
// Each wave owns a 16-row query tile with online softmax.
// ---------------------------------------------------------------------------
__global__ __launch_bounds__(256) void attn_kernel(const _Float16* __restrict__ qkv,
                                                   _Float16* __restrict__ attn_out) {
  __shared__ __align__(16) _Float16 Ksh[32 * DD];      // 4KB  [key][d]
  __shared__ __align__(16) _Float16 Vt[DD * 32];       // 4KB  [d][key]
  __shared__ __align__(16) _Float16 Psh[8][16 * 32];   // 8KB  per-wave P tiles

  int tid = threadIdx.x;
  int wave = tid >> 5;
  int l = tid & 31;
  int col = l & 15, half = l >> 4;

  int bh = blockIdx.y;
  int b = bh >> 3, h = bh & (NH - 1);
  int q0 = (blockIdx.x * 8 + wave) * 16;

  const size_t plane = (size_t)NN * DD;
  const _Float16* Q = qkv + (((size_t)0 * BB + b) * NH + h) * plane;
  const _Float16* K = qkv + (((size_t)1 * BB + b) * NH + h) * plane;
  const _Float16* V = qkv + (((size_t)2 * BB + b) * NH + h) * plane;

  // query fragments are key-loop invariant
  v16h qa0 = load_frag_a(Q + (size_t)q0 * DD, DD);
  v16h qa1 = load_frag_a(Q + (size_t)q0 * DD + 32, DD);

  float mrow[8], lrow[8];
#pragma unroll
  for (int j = 0; j < 8; ++j) { mrow[j] = -3.0e38f; lrow[j] = 0.0f; }
  v8f acc0 = {}, acc1 = {}, acc2 = {}, acc3 = {};

  const float scale = 0.125f;            // d^-0.5, d=64
  const int vkey = tid >> 3;             // staging: one key row per 8 threads
  const int vdg  = (tid & 7) * 8;        // 8 d-elements per thread

  for (int kb = 0; kb < NN / 32; ++kb) {
    int k0 = kb * 32;

    // read V row chunk to registers (global, coalesced) before touching LDS
    union { uint4 q; _Float16 hh[8]; } vu;
    vu.q = *(const uint4*)(V + (size_t)(k0 + vkey) * DD + vdg);

    __syncthreads();   // previous tile fully consumed

    // V: transposed scatter into LDS
#pragma unroll
    for (int i = 0; i < 8; ++i) Vt[(vdg + i) * 32 + vkey] = vu.hh[i];

    // K: TDM 2-D tile DMA (4KB) issued by wave 0; fallback = cooperative copy
#if USE_TDM
    if (wave == 0) {
      tdm_load_2d_f16((unsigned)(size_t)&Ksh[0],
                      K + (size_t)k0 * DD,
                      /*tensor_d0=*/DD, /*tensor_d1=*/NN,
                      /*tile_d0=*/DD, /*tile_d1=*/32,
                      /*stride0=*/DD);
      __builtin_amdgcn_s_wait_tensorcnt(0);
    }
#else
    *(uint4*)&Ksh[vkey * DD + vdg] = *(const uint4*)(K + (size_t)(k0 + vkey) * DD + vdg);
#endif
    __syncthreads();   // tiles published

    // S = Q K^T from LDS K tile: left keys 0..15, right keys 16..31
    v8f sL = {}, sR = {};
    sL = wmma16x16x32(qa0, load_frag_b_kcontig(Ksh, DD), sL);
    sL = wmma16x16x32(qa1, load_frag_b_kcontig(Ksh + 32, DD), sL);
    sR = wmma16x16x32(qa0, load_frag_b_kcontig(Ksh + 16 * DD, DD), sR);
    sR = wmma16x16x32(qa1, load_frag_b_kcontig(Ksh + 16 * DD + 32, DD), sR);

#pragma unroll
    for (int j = 0; j < 8; ++j) {
      float a = sL[j] * scale;
      float c = sR[j] * scale;
      float mx = fmaxf(a, c);
      mx = fmaxf(mx, __shfl_xor(mx, 1));
      mx = fmaxf(mx, __shfl_xor(mx, 2));
      mx = fmaxf(mx, __shfl_xor(mx, 4));
      mx = fmaxf(mx, __shfl_xor(mx, 8));   // stays inside each 16-lane half
      float mnew = fmaxf(mrow[j], mx);
      float corr = __expf(mrow[j] - mnew);
      float eL = __expf(a - mnew);
      float eR = __expf(c - mnew);
      float rs = eL + eR;
      rs += __shfl_xor(rs, 1);
      rs += __shfl_xor(rs, 2);
      rs += __shfl_xor(rs, 4);
      rs += __shfl_xor(rs, 8);
      lrow[j] = lrow[j] * corr + rs;
      mrow[j] = mnew;
      acc0[j] *= corr; acc1[j] *= corr; acc2[j] *= corr; acc3[j] *= corr;
      int row = j + 8 * half;
      Psh[wave][row * 32 + col]      = (_Float16)eL;
      Psh[wave][row * 32 + 16 + col] = (_Float16)eR;
    }
    // same-wave LDS store->load is in-order; no barrier needed for Psh
    v16h pa = load_frag_a(&Psh[wave][0], 32);
    acc0 = wmma16x16x32(pa, load_frag_b_kcontig(Vt + 0 * 16 * 32, 32), acc0);
    acc1 = wmma16x16x32(pa, load_frag_b_kcontig(Vt + 1 * 16 * 32, 32), acc1);
    acc2 = wmma16x16x32(pa, load_frag_b_kcontig(Vt + 2 * 16 * 32, 32), acc2);
    acc3 = wmma16x16x32(pa, load_frag_b_kcontig(Vt + 3 * 16 * 32, 32), acc3);
  }

  // finalize: O / l, write [B,N,C] f16 at channel h*64 + t*16 + col
#pragma unroll
  for (int j = 0; j < 8; ++j) {
    float inv = 1.0f / lrow[j];
    int n = q0 + j + 8 * half;
    size_t base = ((size_t)b * NN + n) * CC + h * DD + col;
    attn_out[base]      = (_Float16)(acc0[j] * inv);
    attn_out[base + 16] = (_Float16)(acc1[j] * inv);
    attn_out[base + 32] = (_Float16)(acc2[j] * inv);
    attn_out[base + 48] = (_Float16)(acc3[j] * inv);
  }
}

// ---------------------------------------------------------------------------
// K5: residual add + LayerNorm2 + transpose back to [B,C,H,W] f32.
// ---------------------------------------------------------------------------
__global__ __launch_bounds__(256) void ln2_kernel(const float* __restrict__ x,
                                                  const float* __restrict__ proj,
                                                  const float* __restrict__ w,
                                                  const float* __restrict__ bias,
                                                  float* __restrict__ out) {
  int bn = blockIdx.x;
  int b = bn >> 10, n = bn & (NN - 1);
  int tid = threadIdx.x;
  const float* xp = x + (size_t)b * CC * NN + n;
  const float* pp = proj + ((size_t)b * NN + n) * CC;

  float v0 = xp[(size_t)tid * NN]         + pp[tid];
  float v1 = xp[(size_t)(tid + 256) * NN] + pp[tid + 256];

  __shared__ float s1[256], s2[256];
  s1[tid] = v0 + v1;
  s2[tid] = v0 * v0 + v1 * v1;
  __syncthreads();
#pragma unroll
  for (int off = 128; off > 0; off >>= 1) {
    if (tid < off) { s1[tid] += s1[tid + off]; s2[tid] += s2[tid + off]; }
    __syncthreads();
  }
  float mean = s1[0] * (1.0f / CC);
  float var  = s2[0] * (1.0f / CC) - mean * mean;
  float rstd = rsqrtf(var + 1e-5f);

  float* op = out + (size_t)b * CC * NN + n;
  op[(size_t)tid * NN]         = (v0 - mean) * rstd * w[tid]       + bias[tid];
  op[(size_t)(tid + 256) * NN] = (v1 - mean) * rstd * w[tid + 256] + bias[tid + 256];
}

// ---------------------------------------------------------------------------
// Launcher
// ---------------------------------------------------------------------------
extern "C" void kernel_launch(void* const* d_in, const int* in_sizes, int n_in,
                              void* d_out, int out_size, void* d_ws, size_t ws_size,
                              hipStream_t stream) {
  (void)in_sizes; (void)n_in; (void)out_size; (void)ws_size;

  const float* x     = (const float*)d_in[0];
  const float* ln1_w = (const float*)d_in[1];
  const float* ln1_b = (const float*)d_in[2];
  const float* qkv_w = (const float*)d_in[3];
  const float* qkv_b = (const float*)d_in[4];
  const float* out_w = (const float*)d_in[5];
  const float* out_b = (const float*)d_in[6];
  const float* ln2_w = (const float*)d_in[7];
  const float* ln2_b = (const float*)d_in[8];
  float* out = (float*)d_out;

  _Float16* qkvw_h = (_Float16*)d_ws;                         // 1536*512   f16
  _Float16* outw_h = qkvw_h + (size_t)C3 * CC;                // 512*512    f16
  _Float16* xn_h   = outw_h + (size_t)CC * CC;                // B*N*C      f16
  _Float16* qkv_h  = xn_h   + (size_t)BB * NN * CC;           // 3*B*NH*N*D f16
  _Float16* attn_h = qkv_h  + (size_t)3 * BB * NH * NN * DD;  // B*N*C      f16
  float*    proj_f = (float*)(attn_h + (size_t)BB * NN * CC); // B*N*C      f32

  cvt_f16_kernel<<<(C3 * CC + 255) / 256, 256, 0, stream>>>(qkv_w, qkvw_h, C3 * CC);
  cvt_f16_kernel<<<(CC * CC + 255) / 256, 256, 0, stream>>>(out_w, outw_h, CC * CC);

  ln1_kernel<<<BB * NN, 256, 0, stream>>>(x, ln1_w, ln1_b, xn_h);

  qkv_gemm_kernel<<<dim3(C3 / 128, NN / 16, BB), 256, 0, stream>>>(xn_h, qkvw_h, qkv_b, qkv_h);

  attn_kernel<<<dim3(NN / 128, BB * NH), 256, 0, stream>>>(qkv_h, attn_h);

  out_gemm_kernel<<<dim3(CC / 128, NN / 16, BB), 256, 0, stream>>>(attn_h, outw_h, out_b, proj_f);

  ln2_kernel<<<BB * NN, 256, 0, stream>>>(x, proj_f, ln2_w, ln2_b, out);
}
